// MambaBlock_42640435314848
// MI455X (gfx1250) — matrix operand order repaired and checked by reading, not compiled, not run on hardware
//
#include <hip/hip_runtime.h>
#include <hip/hip_bf16.h>
#include <cstddef>

// ---------------------------------------------------------------------------
// MambaBlock for MI455X (gfx1250, wave32, WMMA).
//   DIM=128, DSTATE=256, B=64, T=4096, fp32 reference.
// Strategy:
//   k_cvt      : fp32 weights -> bf16 copies in workspace (once per launch)
//   k_gemm_in  : xp = x @ W_in^T + b_in            (WMMA bf16, fp32 accum)
//   k_scan     : sequential recurrence, 4 WGs x 16 waves, weights in VGPRs,
//                xt software-pipelined (register prefetch) off the critical path
//   k_gemm_out : out = (y + xp @ W_d^T + b_d) @ W_out^T + b_out (fused)
// ---------------------------------------------------------------------------

#define DIM     128
#define DSTATE  256
#define KAB     384   // DSTATE + DIM
#define TLEN    4096
#define BATCH   64
#define NROWS   (BATCH * TLEN)          // 262144
#define NTOT    ((size_t)NROWS * DIM)   // 33554432 floats

typedef __bf16 bf16_t;
typedef __attribute__((ext_vector_type(16))) __bf16 v16bf;
typedef __attribute__((ext_vector_type(8)))  __bf16 v8bf;
typedef __attribute__((ext_vector_type(4)))  __bf16 v4bf;
typedef __attribute__((ext_vector_type(8)))  float  v8f;
typedef __attribute__((ext_vector_type(4)))  float  v4f;

// ---- CDNA5 bf16 WMMA fragment loaders (documented VGPR layouts) -----------

// A-matrix 16x32 bf16 fragment from row-major bf16 (LDS or global).
// lane L: row M = L%16; lanes 0-15 hold K = kbase+[0..7] and kbase+[16..23],
// lanes 16-31 hold K = kbase+[8..15] and kbase+[24..31].
__device__ __forceinline__ v16bf ld_afrag_bf16(const bf16_t* base, int ldE,
                                               int kbase, int lane) {
  const int m  = lane & 15;
  const int hs = ((lane >> 4) & 1) * 8;
  const bf16_t* p = base + m * ldE + kbase + hs;
  v8bf lo = *(const v8bf*)(p);
  v8bf hi = *(const v8bf*)(p + 16);
  v16bf r;
#pragma unroll
  for (int i = 0; i < 8; ++i) { r[i] = lo[i]; r[i + 8] = hi[i]; }
  return r;
}

// Same A fragment but sourced from row-major fp32, converted to bf16.
__device__ __forceinline__ v16bf ld_afrag_f32(const float* X, int ldE,
                                              int row0, int kbase, int lane) {
  const int m  = row0 + (lane & 15);
  const int hs = ((lane >> 4) & 1) * 8;
  const float* p = X + (size_t)m * ldE + kbase + hs;
  v4f a0 = *(const v4f*)(p);
  v4f a1 = *(const v4f*)(p + 4);
  v4f b0 = *(const v4f*)(p + 16);
  v4f b1 = *(const v4f*)(p + 20);
  v16bf r;
#pragma unroll
  for (int i = 0; i < 4; ++i) {
    r[i]      = (bf16_t)a0[i];
    r[i + 4]  = (bf16_t)a1[i];
    r[i + 8]  = (bf16_t)b0[i];
    r[i + 12] = (bf16_t)b1[i];
  }
  return r;
}

// B-matrix 32x16 bf16 fragment, B = W^T where W is [N][K] row-major bf16.
// lane L: column N = n0 + L%16; lanes 0-15 hold K = kbase+[0..7] and kbase+[16..23],
// lanes 16-31 hold K = kbase+[8..15] and kbase+[24..31].
__device__ __forceinline__ v16bf ld_bfrag(const bf16_t* W, int K,
                                          int n0, int kbase, int lane) {
  const int n  = n0 + (lane & 15);
  const int ks = ((lane >> 4) & 1) * 8;
  const bf16_t* p = W + (size_t)n * K + kbase + ks;
  v8bf a = *(const v8bf*)(p);
  v8bf b = *(const v8bf*)(p + 16);
  v16bf r;
#pragma unroll
  for (int i = 0; i < 8; ++i) { r[i] = a[i]; r[i + 8] = b[i]; }
  return r;
}

__device__ __forceinline__ v8f bias_acc(float b) {
  v8f c;
#pragma unroll
  for (int i = 0; i < 8; ++i) c[i] = b;
  return c;
}

#define WMMA_BF16(a, b, c) \
  __builtin_amdgcn_wmma_f32_16x16x32_bf16(false, (a), false, (b), (short)0, (c), false, false)

// ---------------------------------------------------------------------------
// Kernel 0: fp32 -> bf16 weight conversion
// ---------------------------------------------------------------------------
__global__ void k_cvt(const float* __restrict__ src, bf16_t* __restrict__ dst, int n) {
  int i = blockIdx.x * blockDim.x + threadIdx.x;
  if (i < n) dst[i] = (bf16_t)src[i];
}

// ---------------------------------------------------------------------------
// Kernel 1: xp = x @ W_in^T + b_in      (rows tiled by 16, 8 waves = 8 N-tiles)
// ---------------------------------------------------------------------------
__global__ __launch_bounds__(256) void k_gemm_in(
    const float* __restrict__ x, const bf16_t* __restrict__ Win,
    const float* __restrict__ b_in, float* __restrict__ xp) {
  const int lane = threadIdx.x & 31;
  const int wave = threadIdx.x >> 5;       // N-tile 0..7
  const int row0 = blockIdx.x * 16;

  v16bf bW[4];
#pragma unroll
  for (int ks = 0; ks < 4; ++ks) bW[ks] = ld_bfrag(Win, DIM, wave * 16, ks * 32, lane);

  v8f acc = bias_acc(b_in[wave * 16 + (lane & 15)]);
#pragma unroll
  for (int ks = 0; ks < 4; ++ks) {
    v16bf a = ld_afrag_f32(x, DIM, row0, ks * 32, lane);
    acc = WMMA_BF16(a, bW[ks], acc);
  }
  const int n  = wave * 16 + (lane & 15);
  const int mo = ((lane >> 4) & 1) * 8;
#pragma unroll
  for (int r = 0; r < 8; ++r)
    xp[(size_t)(row0 + mo + r) * DIM + n] = acc[r];
}

// ---------------------------------------------------------------------------
// Kernel 2: the sequential scan. grid = 4 (16 batches each), block = 512.
// Weights stay in VGPRs across all 4096 steps; latent/h live in LDS.
// xt is double-buffered through registers so the per-step critical path never
// waits on global memory; a deep prefetch keeps xp lines warm.
// ---------------------------------------------------------------------------
#define ALD 392   // bf16 A-tile row stride  (16B aligned, cols 0..255=h, 256..383=xt)
#define HLD 264   // fp32 h-tile row stride

__global__ __launch_bounds__(512) void k_scan(
    const float* __restrict__ xp, const bf16_t* __restrict__ Wab,
    const float* __restrict__ b_ab, const float* __restrict__ w_ln,
    const bf16_t* __restrict__ Wc, const float* __restrict__ b_c,
    float* __restrict__ y) {
  __shared__ bf16_t aLds[16 * ALD];
  __shared__ float  hLds[16 * HLD];

  const int lane = threadIdx.x & 31;
  const int wave = threadIdx.x >> 5;       // 0..15, N-tile of DSTATE
  const int b0   = blockIdx.x * 16;        // batch rows owned by this WG

  // Persistent register-resident B fragments.
  v16bf bAB[12];
#pragma unroll
  for (int ks = 0; ks < 12; ++ks) bAB[ks] = ld_bfrag(Wab, KAB, wave * 16, ks * 32, lane);
  const float babv = b_ab[wave * 16 + (lane & 15)];

  v16bf bC[8];
  float bcv = 0.f;
  if (wave < 8) {
#pragma unroll
    for (int ks = 0; ks < 8; ++ks) bC[ks] = ld_bfrag(Wc, DSTATE, wave * 16, ks * 32, lane);
    bcv = b_c[wave * 16 + (lane & 15)];
  }

  float wlnv[8];
#pragma unroll
  for (int i = 0; i < 8; ++i) wlnv[i] = w_ln[lane + 32 * i];

  // Per-thread slice of the 16x128 xt tile: 4 consecutive floats.
  const int xi = threadIdx.x * 4;          // 512 threads x 4 = 2048 = 16*128
  const int xr = xi >> 7;                  // row 0..15
  const int xc = xi & 127;                 // col 0..124
  const float* xpBase = xp + ((size_t)(b0 + xr) * TLEN) * DIM + xc;
  bf16_t* xLds = &aLds[xr * ALD + 256 + xc];

  // latent(0) = 0
  for (int i = threadIdx.x; i < 16 * 256; i += 512) {
    aLds[(i >> 8) * ALD + (i & 255)] = (bf16_t)0.f;
  }

  // Prologue: prefetch xt(0) into registers.
  v4f xreg = *(const v4f*)(xpBase);
  __syncthreads();

  for (int t = 0; t < TLEN; ++t) {
    // Commit prefetched xt(t) to LDS (single 8-byte packed store), then
    // immediately issue the load for xt(t+1); its latency is hidden behind
    // this step's WMMA chain + RMSNorm.
    {
      v4bf pk;
      pk[0] = (bf16_t)xreg[0]; pk[1] = (bf16_t)xreg[1];
      pk[2] = (bf16_t)xreg[2]; pk[3] = (bf16_t)xreg[3];
      *(v4bf*)xLds = pk;
      const int tn = (t + 1 < TLEN) ? (t + 1) : t;     // clamp, stays in-bounds
      xreg = *(const v4f*)(xpBase + (size_t)tn * DIM);
      // Deep prefetch: keep xp(t+16) warm in cache.
      if (t + 16 < TLEN)
        __builtin_prefetch(xpBase + (size_t)(t + 16) * DIM, 0, 1);
    }
    __syncthreads();

    // h = [latent ‖ xt] @ W_ab^T + b_ab : 12 dependent WMMAs per wave.
    v8f acc = bias_acc(babv);
#pragma unroll
    for (int ks = 0; ks < 12; ++ks) {
      v16bf a = ld_afrag_bf16(aLds, ALD, ks * 32, lane);
      acc = WMMA_BF16(a, bAB[ks], acc);
    }
    {
      const int n  = wave * 16 + (lane & 15);
      const int mo = ((lane >> 4) & 1) * 8;
#pragma unroll
      for (int r = 0; r < 8; ++r) hLds[(mo + r) * HLD + n] = acc[r];
    }
    __syncthreads();

    // RMSNorm row `wave` (fp32), write normalized bf16 h back as next latent
    // and as the A-matrix for the W_c matmul.
    {
      float v[8]; float ss = 0.f;
#pragma unroll
      for (int i = 0; i < 8; ++i) {
        v[i] = hLds[wave * HLD + lane + 32 * i];
        ss += v[i] * v[i];
      }
#pragma unroll
      for (int off = 16; off > 0; off >>= 1) ss += __shfl_xor(ss, off, 32);
      const float rs = rsqrtf(ss * (1.0f / (float)DSTATE) + 1e-5f);
#pragma unroll
      for (int i = 0; i < 8; ++i)
        aLds[wave * ALD + lane + 32 * i] = (bf16_t)(v[i] * rs * wlnv[i]);
    }
    __syncthreads();

    // y = h @ W_c^T + b_c : waves 0..7 (8 N-tiles of DIM), 8 WMMAs each.
    if (wave < 8) {
      v8f ya = bias_acc(bcv);
#pragma unroll
      for (int ks = 0; ks < 8; ++ks) {
        v16bf a = ld_afrag_bf16(aLds, ALD, ks * 32, lane);
        ya = WMMA_BF16(a, bC[ks], ya);
      }
      const int n  = wave * 16 + (lane & 15);
      const int mo = ((lane >> 4) & 1) * 8;
#pragma unroll
      for (int r = 0; r < 8; ++r)
        y[((size_t)(b0 + mo + r) * TLEN + t) * DIM + n] = ya[r];
    }
  }
}

// ---------------------------------------------------------------------------
// Kernel 3: out = (y + xp @ W_d^T + b_d) @ W_out^T + b_out   (y in d_out, in place)
// ---------------------------------------------------------------------------
#define TLD 136   // bf16 staging tile row stride (16B aligned)

__global__ __launch_bounds__(256) void k_gemm_out(
    const float* __restrict__ xp, const bf16_t* __restrict__ Wd,
    const float* __restrict__ b_d, const bf16_t* __restrict__ Wout,
    const float* __restrict__ b_out, float* __restrict__ out) {
  __shared__ bf16_t tLds[16 * TLD];
  const int lane = threadIdx.x & 31;
  const int wave = threadIdx.x >> 5;
  const int row0 = blockIdx.x * 16;
  const int n    = wave * 16 + (lane & 15);
  const int mo   = ((lane >> 4) & 1) * 8;

  v16bf bD[4], bO[4];
#pragma unroll
  for (int ks = 0; ks < 4; ++ks) {
    bD[ks] = ld_bfrag(Wd,   DIM, wave * 16, ks * 32, lane);
    bO[ks] = ld_bfrag(Wout, DIM, wave * 16, ks * 32, lane);
  }

  // d = xp @ W_d^T + b_d + y
  v8f acc = bias_acc(b_d[n]);
#pragma unroll
  for (int ks = 0; ks < 4; ++ks) {
    v16bf a = ld_afrag_f32(xp, DIM, row0, ks * 32, lane);
    acc = WMMA_BF16(a, bD[ks], acc);
  }
#pragma unroll
  for (int r = 0; r < 8; ++r) {
    acc[r] += out[(size_t)(row0 + mo + r) * DIM + n];
    tLds[(mo + r) * TLD + n] = (bf16_t)acc[r];
  }
  __syncthreads();

  // out = d @ W_out^T + b_out
  v8f acc2 = bias_acc(b_out[n]);
#pragma unroll
  for (int ks = 0; ks < 4; ++ks) {
    v16bf a = ld_afrag_bf16(tLds, TLD, ks * 32, lane);
    acc2 = WMMA_BF16(a, bO[ks], acc2);
  }
#pragma unroll
  for (int r = 0; r < 8; ++r)
    out[(size_t)(row0 + mo + r) * DIM + n] = acc2[r];
}

// ---------------------------------------------------------------------------
extern "C" void kernel_launch(void* const* d_in, const int* in_sizes, int n_in,
                              void* d_out, int out_size, void* d_ws, size_t ws_size,
                              hipStream_t stream) {
  (void)in_sizes; (void)n_in; (void)out_size; (void)ws_size;
  const float* x     = (const float*)d_in[0];
  const float* W_in  = (const float*)d_in[1];
  const float* b_in  = (const float*)d_in[2];
  const float* W_ab  = (const float*)d_in[3];
  const float* b_ab  = (const float*)d_in[4];
  const float* w_ln  = (const float*)d_in[5];
  const float* W_c   = (const float*)d_in[6];
  const float* b_c   = (const float*)d_in[7];
  const float* W_d   = (const float*)d_in[8];
  const float* b_d   = (const float*)d_in[9];
  const float* W_out = (const float*)d_in[10];
  const float* b_out = (const float*)d_in[11];
  float* out = (float*)d_out;

  char* ws = (char*)d_ws;
  float*  xp     = (float*)ws;                      size_t off = NTOT * sizeof(float);
  bf16_t* Wab_b  = (bf16_t*)(ws + off);             off += (size_t)DSTATE * KAB * 2;
  bf16_t* Wc_b   = (bf16_t*)(ws + off);             off += (size_t)DIM * DSTATE * 2;
  bf16_t* Win_b  = (bf16_t*)(ws + off);             off += (size_t)DIM * DIM * 2;
  bf16_t* Wd_b   = (bf16_t*)(ws + off);             off += (size_t)DIM * DIM * 2;
  bf16_t* Wout_b = (bf16_t*)(ws + off);             off += (size_t)DIM * DIM * 2;

  // Weight conversions (tiny).
  k_cvt<<<(DSTATE * KAB + 255) / 256, 256, 0, stream>>>(W_ab,  Wab_b,  DSTATE * KAB);
  k_cvt<<<(DIM * DSTATE + 255) / 256, 256, 0, stream>>>(W_c,   Wc_b,   DIM * DSTATE);
  k_cvt<<<(DIM * DIM + 255) / 256,    256, 0, stream>>>(W_in,  Win_b,  DIM * DIM);
  k_cvt<<<(DIM * DIM + 255) / 256,    256, 0, stream>>>(W_d,   Wd_b,   DIM * DIM);
  k_cvt<<<(DIM * DIM + 255) / 256,    256, 0, stream>>>(W_out, Wout_b, DIM * DIM);

  // xp = x @ W_in^T + b_in
  k_gemm_in<<<NROWS / 16, 256, 0, stream>>>(x, Win_b, b_in, xp);

  // Sequential recurrence -> y written into d_out.
  k_scan<<<BATCH / 16, 512, 0, stream>>>(xp, Wab_b, b_ab, w_ln, Wc_b, b_c, out);

  // out = (y + xp @ W_d^T + b_d) @ W_out^T + b_out   (in place on d_out)
  k_gemm_out<<<NROWS / 16, 256, 0, stream>>>(xp, Wd_b, b_d, Wout_b, b_out, out);
}
